// QuantumKANLayer_34291018891677
// MI455X (gfx1250) — matrix operand order repaired and checked by reading, not compiled, not run on hardware
//
#include <hip/hip_runtime.h>
#include <hip/hip_bf16.h>

// Types for CDNA5 WMMA intrinsics
typedef __attribute__((ext_vector_type(16))) __bf16          v16bf;
typedef __attribute__((ext_vector_type(8)))  float           v8f;
typedef __attribute__((ext_vector_type(16))) unsigned short  v16u;

#define BATCH  32768
#define N_IN   256
#define N_OUT  256
#define KHARM  8
#define KTOT   (N_IN * KHARM * 2)   // 4096 = GEMM K dimension
#define M_TILE 64
#define KB_CNT (KTOT / 32)          // 128 K-steps of 32 (even)

static __device__ __forceinline__ unsigned short f2bf(float f) {
    unsigned u = __builtin_bit_cast(unsigned, f);
    u += 0x7FFFu + ((u >> 16) & 1u);   // round-to-nearest-even
    return (unsigned short)(u >> 16);
}

// ---------------------------------------------------------------------------
// Pack W = interleave(coef_a, coef_b) into bf16 B-fragments.
// Fragment (kb, nb) = 32x16 bf16 tile of W for K-block kb, N-block nb.
// Per-lane layout (ISA B 32x16, 16-bit): lane L, slot s (0..15):
//   K = (s>=8 ? 16 : 0) + (L>=16 ? 8 : 0) + (s&7),  N = nb*16 + (L&15)
// Flat index: (((kb*16 + nb)*32 + lane)*16 + slot)
// K decode (must match A generation below):
//   i = kb*2 + (K>>4); t = K&15; harmonic h = t>>1; trig = t&1 (0=cos->a, 1=sin->b)
// ---------------------------------------------------------------------------
__global__ __launch_bounds__(256) void kan_pack_w(const float* __restrict__ ca,
                                                  const float* __restrict__ cb,
                                                  unsigned short* __restrict__ wp) {
    int idx  = blockIdx.x * 256 + threadIdx.x;     // 0 .. 1048575
    int f    = idx >> 9;                           // fragment = kb*16 + nb
    int rem  = idx & 511;
    int lane = rem >> 4;
    int slot = rem & 15;
    int kb   = f >> 4;
    int nb   = f & 15;
    int K    = ((slot >= 8) ? 16 : 0) + ((lane >= 16) ? 8 : 0) + (slot & 7);
    int n    = nb * 16 + (lane & 15);
    int i    = kb * 2 + (K >> 4);
    int t    = K & 15;
    int h    = t >> 1;
    const float* src = (t & 1) ? cb : ca;          // sin -> coef_b, cos -> coef_a
    float v = src[((size_t)n * N_IN + i) * KHARM + h];
    wp[idx] = f2bf(v);
}

// Branch-free: produce the lane's 4 harmonics cos/sin(m*t), m = base+1..base+4,
// where base = 0 (lanes 0-15) or 4 (lanes 16-31). Start point (c1,s1) or (c5,s5)
// obtained via two doublings + one addition, selected without divergence.
static __device__ __forceinline__ void gen4(float t, bool hi,
                                            float* __restrict__ C,
                                            float* __restrict__ S) {
    float s1, c1;
    __sincosf(t, &s1, &c1);
    const float c2 = c1 * c1 - s1 * s1;
    const float s2 = 2.0f * s1 * c1;
    const float c4 = c2 * c2 - s2 * s2;
    const float s4 = 2.0f * s2 * c2;
    const float c5 = c4 * c1 - s4 * s1;
    const float s5 = s4 * c1 + c4 * s1;
    float c = hi ? c5 : c1;
    float s = hi ? s5 : s1;
#pragma unroll
    for (int q = 0; q < 4; ++q) {
        C[q] = c; S[q] = s;
        const float cn = c * c1 - s * s1;
        const float sn = s * c1 + c * s1;
        c = cn; s = sn;
    }
}

// ---------------------------------------------------------------------------
// Main GEMM: block = 256 threads (8 waves), computes 64(M) x 256(N).
// Waves: 4 in M x 2 in N. Each wave: 1 A fragment band (16 rows),
// 8 accumulators covering 128 columns. 8 WMMA per K-step.
// B fragments ping-pong double-buffered; x-phase LDS reads software-pipelined
// one half-iteration ahead so ds latency hides under WMMA issue.
// ---------------------------------------------------------------------------
__global__ __launch_bounds__(256) void kan_wmma_gemm(const float* __restrict__ x,
                                                     const unsigned short* __restrict__ wp,
                                                     float* __restrict__ out) {
    // x tile transposed in LDS with XOR swizzle: xs[i*64 + (row ^ (i&63))]
    __shared__ float xs[N_IN * M_TILE];   // 64 KB

    const int    tid    = threadIdx.x;
    const size_t rowBlk = (size_t)blockIdx.x * M_TILE;

    // ---- stage x[rowBlk .. rowBlk+63][0..255] into LDS (swizzled transpose)
#pragma unroll
    for (int it = 0; it < 16; ++it) {
        int flat = it * 256 + tid;              // float4 units: 64 per row
        int r    = flat >> 6;
        int c4   = (flat & 63) * 4;
        const float4 v = ((const float4*)(x + (rowBlk + r) * N_IN))[c4 >> 2];
        xs[(c4 + 0) * 64 + (r ^ ((c4 + 0) & 63))] = v.x;
        xs[(c4 + 1) * 64 + (r ^ ((c4 + 1) & 63))] = v.y;
        xs[(c4 + 2) * 64 + (r ^ ((c4 + 2) & 63))] = v.z;
        xs[(c4 + 3) * 64 + (r ^ ((c4 + 3) & 63))] = v.w;
    }
    __syncthreads();

    const int  lane      = tid & 31;
    const int  wave      = tid >> 5;
    const int  waveM     = wave >> 1;            // 0..3
    const int  waveN     = wave & 1;             // 0..1
    const int  rowInTile = waveM * 16 + (lane & 15);
    const bool hiLane    = (lane >= 16);         // harmonics 5-8 vs 1-4

    v8f acc[8];
#pragma unroll
    for (int nt = 0; nt < 8; ++nt) acc[nt] = (v8f){0.f,0.f,0.f,0.f,0.f,0.f,0.f,0.f};

    // B fragment pointer: fragment (kb, nt) for this wave's column half
    auto bptr = [&](int kb, int nt) {
        return (const v16u*)(wp +
            ((size_t)((kb * 16 + waveN * 8 + nt) * 32 + lane)) * 16);
    };

    // swizzled x-phase read for input feature i
    auto ldx = [&](int i) { return xs[i * 64 + (rowInTile ^ (i & 63))]; };

    // Build the A fragment (ISA 16-bit A 16x32 layout) from the two phases:
    // slots 0-7 -> K = base+s (feature i0), slots 8-15 -> K = base+16+(s-8) (i1)
    auto makeA = [&](float t0, float t1) {
        float C0[4], S0[4], C1[4], S1[4];
        gen4(t0, hiLane, C0, S0);
        gen4(t1, hiLane, C1, S1);
        v16bf A;
#pragma unroll
        for (int q = 0; q < 4; ++q) {
            A[2 * q + 0] = (__bf16)C0[q];   // lowers to v_cvt_pk_bf16_f32
            A[2 * q + 1] = (__bf16)S0[q];
            A[8 + 2 * q] = (__bf16)C1[q];
            A[9 + 2 * q] = (__bf16)S1[q];
        }
        return A;
    };

    v16u b0[8], b1[8];
#pragma unroll
    for (int nt = 0; nt < 8; ++nt) b0[nt] = *bptr(0, nt);

    // 8 back-to-back WMMAs sharing A; reuse_a hint on all but the last.
    // Modifier args must be literal immediates -> manual unroll via macro.
#define KAN_WMMA8(BUF, A)                                                 \
    do {                                                                  \
        acc[0] = __builtin_amdgcn_wmma_f32_16x16x32_bf16(false, A, false, \
            __builtin_bit_cast(v16bf, BUF[0]), (short)0, acc[0], true,  false); \
        acc[1] = __builtin_amdgcn_wmma_f32_16x16x32_bf16(false, A, false, \
            __builtin_bit_cast(v16bf, BUF[1]), (short)0, acc[1], true,  false); \
        acc[2] = __builtin_amdgcn_wmma_f32_16x16x32_bf16(false, A, false, \
            __builtin_bit_cast(v16bf, BUF[2]), (short)0, acc[2], true,  false); \
        acc[3] = __builtin_amdgcn_wmma_f32_16x16x32_bf16(false, A, false, \
            __builtin_bit_cast(v16bf, BUF[3]), (short)0, acc[3], true,  false); \
        acc[4] = __builtin_amdgcn_wmma_f32_16x16x32_bf16(false, A, false, \
            __builtin_bit_cast(v16bf, BUF[4]), (short)0, acc[4], true,  false); \
        acc[5] = __builtin_amdgcn_wmma_f32_16x16x32_bf16(false, A, false, \
            __builtin_bit_cast(v16bf, BUF[5]), (short)0, acc[5], true,  false); \
        acc[6] = __builtin_amdgcn_wmma_f32_16x16x32_bf16(false, A, false, \
            __builtin_bit_cast(v16bf, BUF[6]), (short)0, acc[6], true,  false); \
        acc[7] = __builtin_amdgcn_wmma_f32_16x16x32_bf16(false, A, false, \
            __builtin_bit_cast(v16bf, BUF[7]), (short)0, acc[7], false, false); \
    } while (0)

    // x phases for kb=0, issued well before first use
    float t0a = ldx(0), t1a = ldx(1);

    for (int kb = 0; kb < KB_CNT; kb += 2) {
        // refill b1 with kb+1 (always valid: KB_CNT even); in flight during
        // A0 generation and the b0 WMMAs
#pragma unroll
        for (int nt = 0; nt < 8; ++nt) b1[nt] = *bptr(kb + 1, nt);

        // prefetch x phases for kb+1 (consumed after the b0 WMMA block)
        const float t0b = ldx(2 * kb + 2);
        const float t1b = ldx(2 * kb + 3);

        const v16bf A0 = makeA(t0a, t1a);
        KAN_WMMA8(b0, A0);

        // refill b0 with kb+2 and prefetch its x phases; both in flight during
        // A1 generation and the b1 WMMAs
        if (kb + 2 < KB_CNT) {
#pragma unroll
            for (int nt = 0; nt < 8; ++nt) b0[nt] = *bptr(kb + 2, nt);
            t0a = ldx(2 * kb + 4);
            t1a = ldx(2 * kb + 5);
        }

        const v16bf A1 = makeA(t0b, t1b);
        KAN_WMMA8(b1, A1);
    }
#undef KAN_WMMA8

    // ---- store D (f32 C/D layout: VGPR r -> M=r (lanes 0-15) / M=8+r (lanes 16-31))
    const int    colBase = waveN * 128 + (lane & 15);
    const int    rOff    = (lane >> 4) * 8;
    const size_t rowBase = rowBlk + waveM * 16 + rOff;
#pragma unroll
    for (int nt = 0; nt < 8; ++nt) {
#pragma unroll
        for (int r = 0; r < 8; ++r) {
            out[(rowBase + r) * N_OUT + colBase + nt * 16] = acc[nt][r];
        }
    }
}

extern "C" void kernel_launch(void* const* d_in, const int* in_sizes, int n_in,
                              void* d_out, int out_size, void* d_ws, size_t ws_size,
                              hipStream_t stream) {
    const float* x  = (const float*)d_in[0];
    const float* ca = (const float*)d_in[1];
    const float* cb = (const float*)d_in[2];
    float* out = (float*)d_out;
    unsigned short* wp = (unsigned short*)d_ws;   // 4096*256 bf16 = 2 MB packed W

    // 1) pack coefficients into WMMA B-fragment layout (bf16)
    kan_pack_w<<<(KTOT * N_OUT) / 256, 256, 0, stream>>>(ca, cb, wp);

    // 2) fused feature-generation + WMMA GEMM
    kan_wmma_gemm<<<BATCH / M_TILE, 256, 0, stream>>>(x, wp, out);
}
